// LinearGO_36919538876542
// MI455X (gfx1250) — compile-verified, matrix operand order checked by reading, not run on hardware
//
#include <hip/hip_runtime.h>
#include <stdint.h>

#define IN_DIM  8192
#define OUT_DIM 4096
#define BATCH   2048

#define BM 128
#define BN 64
#define BK 32

typedef __attribute__((ext_vector_type(16))) __bf16 v16bf;
typedef __attribute__((ext_vector_type(8)))  float  v8f;
typedef int v4i __attribute__((vector_size(16)));   // matches builtin param type

typedef __attribute__((address_space(1))) v4i* gptr_v4i;   // global int4*
typedef __attribute__((address_space(3))) v4i* lptr_v4i;   // LDS int4*

#if __has_builtin(__builtin_amdgcn_global_load_async_to_lds_b128)
#define HAVE_ASYNC_LDS 1
#else
#define HAVE_ASYNC_LDS 0
#endif

__device__ __forceinline__ void async_cp_b128(const void* g, void* l) {
#if HAVE_ASYNC_LDS
    __builtin_amdgcn_global_load_async_to_lds_b128(
        (gptr_v4i)g, (lptr_v4i)l, 0, 0);
#else
    *(uint4*)l = *(const uint4*)g;
#endif
}

__device__ __forceinline__ void wait_async_3() {
#if HAVE_ASYNC_LDS
#if __has_builtin(__builtin_amdgcn_s_wait_asynccnt)
    __builtin_amdgcn_s_wait_asynccnt(3);
#else
    asm volatile("s_wait_asynccnt 0x3" ::: "memory");
#endif
#endif
}
__device__ __forceinline__ void wait_async_0() {
#if HAVE_ASYNC_LDS
#if __has_builtin(__builtin_amdgcn_s_wait_asynccnt)
    __builtin_amdgcn_s_wait_asynccnt(0);
#else
    asm volatile("s_wait_asynccnt 0x0" ::: "memory");
#endif
#endif
}

__device__ __forceinline__ unsigned short f32_to_bf16_rne(float f) {
    union { float f; uint32_t u; } x; x.f = f;
    uint32_t u = x.u;
    uint32_t r = u + 0x7FFFu + ((u >> 16) & 1u);   // round-to-nearest-even
    return (unsigned short)(r >> 16);
}

// ---------------------------------------------------------------- zero Wt ---
__global__ void zero_u4_kernel(uint4* __restrict__ p, long n4) {
    long i = (long)blockIdx.x * blockDim.x + threadIdx.x;
    uint4 z = {0u, 0u, 0u, 0u};
    for (; i < n4; i += (long)gridDim.x * blockDim.x) p[i] = z;
}

// ------------------------------------------------------ convert x -> bf16 ---
__global__ void cvt_x_kernel(const float4* __restrict__ x,
                             ushort4* __restrict__ xb, long n4) {
    long i = (long)blockIdx.x * blockDim.x + threadIdx.x;
    for (; i < n4; i += (long)gridDim.x * blockDim.x) {
        float4 v = x[i];
        ushort4 o;
        o.x = f32_to_bf16_rne(v.x);
        o.y = f32_to_bf16_rne(v.y);
        o.z = f32_to_bf16_rne(v.z);
        o.w = f32_to_bf16_rne(v.w);
        xb[i] = o;
    }
}

// --------------------------------------- scatter kernel values into Wt^T ----
// Wt is [OUT_DIM][IN_DIM] (N-major) so the GEMM B-tiles are K-contiguous.
__global__ void scatter_kernel(const float* __restrict__ vals,
                               const int* __restrict__ rows,
                               const int* __restrict__ cols,
                               unsigned short* __restrict__ Wt, int nnz) {
    int i = blockIdx.x * blockDim.x + threadIdx.x;
    if (i < nnz) {
        Wt[(size_t)cols[i] * IN_DIM + rows[i]] = f32_to_bf16_rne(vals[i]);
    }
}

// ------------------------------------------------------------ bf16 WMMA GEMM
// out[BATCH][OUT_DIM] = relu(Xb[BATCH][IN_DIM] * Wt^T + bias)
// 256 threads = 8 waves; 4(M) x 2(N) waves; each wave = 32x32 out (2x2 WMMA).
// Double-buffered LDS filled by async global->LDS copies (ASYNCcnt).
__launch_bounds__(256)
__global__ void gemm_bf16_kernel(const unsigned short* __restrict__ Xb,
                                 const unsigned short* __restrict__ Wt,
                                 const float* __restrict__ bias,
                                 float* __restrict__ out) {
    __shared__ __align__(16) unsigned short As[2][BM * BK];   // 2 x 8 KB
    __shared__ __align__(16) unsigned short Bs[2][BN * BK];   // 2 x 4 KB

    const int tid  = threadIdx.x;
    const int lane = tid & 31;
    const int wave = tid >> 5;
    const int wm   = wave & 3;        // 0..3  -> 32-row slab
    const int wn   = wave >> 2;       // 0..1  -> 32-col slab
    const int bm   = blockIdx.y * BM;
    const int bn   = blockIdx.x * BN;

    const int row16 = lane & 15;            // fragment row (m or n)
    const int kb    = (lane >> 4) << 3;     // 0 or 8: per-lane K sub-chunk

    // per-thread staging coordinates (chunk = 8 bf16 = 16B)
    const int ra0 = tid >> 2;                    // A chunk 0 row
    const int ra1 = (tid + 256) >> 2;            // A chunk 1 row
    const int ka  = (tid & 3) << 3;              // k offset within tile
    const int rb  = tid >> 2;                    // B chunk row

    // issue one tile's staging (3 x b128 per thread) into buffer `buf`
    auto stage = [&](int k0, int buf) {
        async_cp_b128(&Xb[(size_t)(bm + ra0) * IN_DIM + k0 + ka],
                      &As[buf][ra0 * BK + ka]);
        async_cp_b128(&Xb[(size_t)(bm + ra1) * IN_DIM + k0 + ka],
                      &As[buf][ra1 * BK + ka]);
        async_cp_b128(&Wt[(size_t)(bn + rb) * IN_DIM + k0 + ka],
                      &Bs[buf][rb * BK + ka]);
    };

    v8f acc[2][2] = {};

    stage(0, 0);                                  // prologue fill of buffer 0

    for (int k0 = 0, it = 0; k0 < IN_DIM; k0 += BK, ++it) {
        const int cur = it & 1;

        if (k0 + BK < IN_DIM) {
            stage(k0 + BK, cur ^ 1);              // overlap next tile's copy
            wait_async_3();                       // 3 outstanding = next tile
        } else {
            wait_async_0();
        }
        __syncthreads();                          // cur buffer visible to all

        // ---- fragments (ISA 16-bit layout: K = {kb..kb+7, 16+kb..16+kb+7})
        union FragU { uint4 q[2]; v16bf v; };
        FragU fa[2], fb[2];
#pragma unroll
        for (int mi = 0; mi < 2; ++mi) {
            int ar = wm * 32 + mi * 16 + row16;
            fa[mi].q[0] = *(const uint4*)&As[cur][ar * BK + kb];
            fa[mi].q[1] = *(const uint4*)&As[cur][ar * BK + 16 + kb];
        }
#pragma unroll
        for (int ni = 0; ni < 2; ++ni) {
            int br = wn * 32 + ni * 16 + row16;
            fb[ni].q[0] = *(const uint4*)&Bs[cur][br * BK + kb];
            fb[ni].q[1] = *(const uint4*)&Bs[cur][br * BK + 16 + kb];
        }

#pragma unroll
        for (int mi = 0; mi < 2; ++mi)
#pragma unroll
            for (int ni = 0; ni < 2; ++ni)
                acc[mi][ni] = __builtin_amdgcn_wmma_f32_16x16x32_bf16(
                    /*neg_a=*/false, fa[mi].v,
                    /*neg_b=*/false, fb[ni].v,
                    /*c_mod=*/(short)0, acc[mi][ni],
                    /*reuse_a=*/false, /*reuse_b=*/false);

        __syncthreads();     // everyone done reading cur before it is refilled
    }

    // ---- epilogue: C/D layout  m = (lane>>4)*8 + v, n = lane&15 ------------
    const int nloc  = lane & 15;
    const int mhalf = (lane >> 4) << 3;
#pragma unroll
    for (int mi = 0; mi < 2; ++mi) {
#pragma unroll
        for (int ni = 0; ni < 2; ++ni) {
            int gn = bn + wn * 32 + ni * 16 + nloc;
            float bb = bias[gn];
#pragma unroll
            for (int v = 0; v < 8; ++v) {
                int gm = bm + wm * 32 + mi * 16 + mhalf + v;
                float o = acc[mi][ni][v] + bb;
                out[(size_t)gm * OUT_DIM + gn] = o > 0.0f ? o : 0.0f;
            }
        }
    }
}

// ---------------------------------------------------------------------------
extern "C" void kernel_launch(void* const* d_in, const int* in_sizes, int n_in,
                              void* d_out, int out_size, void* d_ws, size_t ws_size,
                              hipStream_t stream) {
    const float* x    = (const float*)d_in[0];   // [BATCH, IN_DIM]
    const float* kv   = (const float*)d_in[1];   // [nnz]
    const float* bias = (const float*)d_in[2];   // [OUT_DIM]
    const int*   rows = (const int*)d_in[3];     // [nnz]
    const int*   cols = (const int*)d_in[4];     // [nnz]
    const int    nnz  = in_sizes[1];

    // workspace: [ Wt bf16 OUT_DIM*IN_DIM (64MB) | Xb bf16 BATCH*IN_DIM (32MB) ]
    unsigned short* Wt = (unsigned short*)d_ws;
    unsigned short* Xb = Wt + (size_t)OUT_DIM * IN_DIM;
    float* out = (float*)d_out;

    // 1) zero Wt (uint4-wide)
    long wt4 = (long)OUT_DIM * IN_DIM / 8;
    zero_u4_kernel<<<(int)((wt4 + 255) / 256), 256, 0, stream>>>((uint4*)Wt, wt4);

    // 2) x -> bf16
    long x4 = (long)BATCH * IN_DIM / 4;
    cvt_x_kernel<<<(int)((x4 + 255) / 256), 256, 0, stream>>>(
        (const float4*)x, (ushort4*)Xb, x4);

    // 3) scatter kernel values into Wt^T
    scatter_kernel<<<(nnz + 255) / 256, 256, 0, stream>>>(kv, rows, cols, Wt, nnz);

    // 4) WMMA GEMM + bias + relu
    dim3 grid(OUT_DIM / BN, BATCH / BM);   // 64 x 16
    gemm_bf16_kernel<<<grid, 256, 0, stream>>>(Xb, Wt, bias, out);
}
